// PatchGCN_88630945120467
// MI455X (gfx1250) — matrix operand order repaired and checked by reading, not compiled, not run on hardware
//
#include <hip/hip_runtime.h>
#include <hip/hip_bf16.h>

typedef __attribute__((ext_vector_type(16))) _Float16 v16h;
typedef __attribute__((ext_vector_type(8)))  _Float16 v8h;
typedef __attribute__((ext_vector_type(4)))  _Float16 v4h;
typedef __attribute__((ext_vector_type(8)))  float    v8f;

namespace {

constexpr int N_NODES = 16384;
constexpr int E_EDGES = 524288;
constexpr int H   = 128;
constexpr int H2  = 256;
constexpr int H4  = 512;
constexpr int DIN = 1024;
constexpr float MSG_EPS = 1e-7f;

// ---------------- order-preserving float<->uint encoding (for atomicMax) ----
__device__ __forceinline__ unsigned f2ord(float f) {
  unsigned u = __float_as_uint(f);
  return (u & 0x80000000u) ? ~u : (u | 0x80000000u);
}
__device__ __forceinline__ float ord2f(unsigned u) {
  unsigned v = (u & 0x80000000u) ? (u & 0x7FFFFFFFu) : ~u;
  return __uint_as_float(v);
}

// ---------------- fills -----------------------------------------------------
__global__ void fill_u32_kernel(unsigned* p, unsigned v, size_t n) {
  size_t i = (size_t)blockIdx.x * blockDim.x + threadIdx.x;
  if (i < n) p[i] = v;
}
__global__ void fill_f32_kernel(float* p, float v, size_t n) {
  size_t i = (size_t)blockIdx.x * blockDim.x + threadIdx.x;
  if (i < n) p[i] = v;
}

// ---------------- WMMA GEMM: C = act(A[MxK] * W[KxN] + bias) ----------------
// Block: 128 thr (4 waves). Block tile 64x64; each wave: 16 rows x 64 cols
// (4 accumulators). A staged row-major f16, B staged TRANSPOSED f16 so all
// fragment reads are contiguous ds_load_b128.
// ACT: 0=none 1=relu 2=tanh 3=sigmoid ; MUL: C = C * act(v)
template<int ACT, bool MUL, int K, int NCOLS>
__global__ __launch_bounds__(128)
void wmma_gemm_kernel(const float* __restrict__ A, const float* __restrict__ W,
                      const float* __restrict__ bias, float* __restrict__ C) {
  const int lane  = threadIdx.x & 31;
  const int wave  = threadIdx.x >> 5;          // 0..3
  const int nBase = blockIdx.x * 64;           // column tile base
  const int mBase = blockIdx.y * 64;           // block row base
  const int mTile = mBase + wave * 16;         // this wave's row tile

  // row stride 40 halves = 80 B (16-byte aligned rows)
  __shared__ _Float16 As[64][40];              // A tile 64x32, row-major
  __shared__ _Float16 Bst[64][40];             // B tile 32x64 stored [col][k]

  v8f acc[4] = {v8f{}, v8f{}, v8f{}, v8f{}};

  const int half = lane >> 4;                  // frag k-half select
  const int rsel = lane & 15;

  for (int k0 = 0; k0 < K; k0 += 32) {
    // ---- stage A 64x32: float4 global loads -> v4h ds_store_b64 ----
#pragma unroll
    for (int it = 0; it < 4; ++it) {
      int chunk = threadIdx.x + it * 128;      // 512 float4-chunks total
      int r  = chunk >> 3;                     // row
      int c4 = chunk & 7;                      // float4 within row
      const float4 f =
          *(const float4*)&A[(size_t)(mBase + r) * K + (k0 + c4 * 4)];
      v4h hh;
      hh[0] = (_Float16)f.x; hh[1] = (_Float16)f.y;
      hh[2] = (_Float16)f.z; hh[3] = (_Float16)f.w;
      *(v4h*)&As[r][c4 * 4] = hh;
    }
    // ---- stage B 32x64 transposed: coalesced column gathers -> b128 stores --
    {
      int c  = threadIdx.x & 63;               // output column
      int rh = threadIdx.x >> 6;               // k half (0..1)
      v8h p0, p1;
#pragma unroll
      for (int r = 0; r < 8; ++r)
        p0[r] = (_Float16)W[(size_t)(k0 + rh * 16 + r) * NCOLS + (nBase + c)];
#pragma unroll
      for (int r = 0; r < 8; ++r)
        p1[r] = (_Float16)W[(size_t)(k0 + rh * 16 + 8 + r) * NCOLS + (nBase + c)];
      *(v8h*)&Bst[c][rh * 16]     = p0;
      *(v8h*)&Bst[c][rh * 16 + 8] = p1;
    }
    __syncthreads();

    if (k0 + 32 < K) {  // emits global_prefetch_b8
      __builtin_prefetch(&A[(size_t)(mBase + (threadIdx.x >> 1)) * K + (k0 + 32)], 0, 3);
    }

    // ---- A fragment: two contiguous 16B chunks (ISA 7.12.2 16-bit A layout)
    v8h a0 = *(const v8h*)&As[rsel][half * 8];
    v8h a1 = *(const v8h*)&As[rsel][half * 8 + 16];
    v16h afrag = __builtin_shufflevector(a0, a1, 0, 1, 2, 3, 4, 5, 6, 7,
                                         8, 9, 10, 11, 12, 13, 14, 15);
    // ---- 4 B fragments + 4 WMMAs re-using afrag ----
#pragma unroll
    for (int j = 0; j < 4; ++j) {
      v8h b0 = *(const v8h*)&Bst[j * 16 + rsel][half * 8];
      v8h b1 = *(const v8h*)&Bst[j * 16 + rsel][half * 8 + 16];
      v16h bfrag = __builtin_shufflevector(b0, b1, 0, 1, 2, 3, 4, 5, 6, 7,
                                           8, 9, 10, 11, 12, 13, 14, 15);
      acc[j] = __builtin_amdgcn_wmma_f32_16x16x32_f16(
          false, afrag, false, bfrag, (short)0, acc[j], false, false);
    }
    __syncthreads();
  }

  // ---- epilogue: C/D layout lane 0..15 rows 0..7, lane 16..31 rows 8..15 --
  const int rbase = half * 8;
#pragma unroll
  for (int j = 0; j < 4; ++j) {
    const int ncol = nBase + j * 16 + rsel;
    const float bv = bias[ncol];
#pragma unroll
    for (int i = 0; i < 8; ++i) {
      float v = acc[j][i] + bv;
      if (ACT == 1)      v = fmaxf(v, 0.0f);
      else if (ACT == 2) v = tanhf(v);
      else if (ACT == 3) v = 1.0f / (1.0f + __expf(-v));
      size_t idx = (size_t)(mTile + rbase + i) * NCOLS + ncol;
      if (MUL) C[idx] = C[idx] * v;
      else     C[idx] = v;
    }
  }
}

// ---------------- edge pass 1: per-(dst,feature) segment max of logits ------
__global__ __launch_bounds__(128)
void edge_max_kernel(const float* __restrict__ x, const int* __restrict__ src,
                     const int* __restrict__ dst, const float* __restrict__ ew,
                     const float* __restrict__ tarr, int layer,
                     unsigned* __restrict__ menc) {
  const int e = blockIdx.x;
  const int h = threadIdx.x;
  const int s = src[e], d = dst[e];
  const float t = tarr[layer];
  float v = fmaxf(x[(size_t)s * H + h] + ew[e], 0.0f) + MSG_EPS;
  atomicMax(&menc[(size_t)d * H + h], f2ord(v * t));
}

// ---------------- edge pass 2: e=exp(logit-m); accumulate sum_e, sum_me -----
__global__ __launch_bounds__(128)
void edge_sum_kernel(const float* __restrict__ x, const int* __restrict__ src,
                     const int* __restrict__ dst, const float* __restrict__ ew,
                     const float* __restrict__ tarr, int layer,
                     const unsigned* __restrict__ menc,
                     float* __restrict__ sum_e, float* __restrict__ sum_me) {
  const int e = blockIdx.x;
  const int h = threadIdx.x;
  const int s = src[e], d = dst[e];
  const float t = tarr[layer];
  float v = fmaxf(x[(size_t)s * H + h] + ew[e], 0.0f) + MSG_EPS;
  float logit = v * t;
  float m = ord2f(menc[(size_t)d * H + h]);
  if (m < -3.0e38f) m = 0.0f;  // jnp.where(isfinite(m), m, 0)
  float ex = __expf(logit - m);
  atomicAdd(&sum_e[(size_t)d * H + h], ex);
  atomicAdd(&sum_me[(size_t)d * H + h], v * ex);
}

// ---------------- agg + root residual: h = sum_me/(sum_e+1e-16) + x ---------
__global__ void agg_kernel(const float* __restrict__ sum_e,
                           const float* __restrict__ sum_me,
                           const float* __restrict__ x,
                           float* __restrict__ out, size_t n) {
  size_t i = (size_t)blockIdx.x * blockDim.x + threadIdx.x;
  if (i < n) out[i] = sum_me[i] / (sum_e[i] + 1e-16f) + x[i];
}

// ---------------- LayerNorm (+relu), optional residual accumulate ----------
template<int C, bool RES>
__global__ void ln_kernel(const float* __restrict__ in, const float* __restrict__ g,
                          const float* __restrict__ b, float* __restrict__ out) {
  __shared__ float red[C];
  const int n = blockIdx.x, j = threadIdx.x;
  float v = in[(size_t)n * C + j];
  red[j] = v;
  __syncthreads();
  for (int s = C / 2; s > 0; s >>= 1) { if (j < s) red[j] += red[j + s]; __syncthreads(); }
  float mu = red[0] / (float)C;
  __syncthreads();
  float dv = v - mu;
  red[j] = dv * dv;
  __syncthreads();
  for (int s = C / 2; s > 0; s >>= 1) { if (j < s) red[j] += red[j + s]; __syncthreads(); }
  float var = red[0] / (float)C;
  float o = dv * rsqrtf(var + 1e-5f) * g[j] + b[j];
  o = fmaxf(o, 0.0f);
  if (RES) out[(size_t)n * C + j] += o;
  else     out[(size_t)n * C + j]  = o;
}

// ---------------- concat slice copy: xcat[:, off:off+H] = x -----------------
__global__ void concat_kernel(const float* __restrict__ x, float* __restrict__ xcat,
                              int off, size_t n) {
  size_t i = (size_t)blockIdx.x * blockDim.x + threadIdx.x;
  if (i < n) {
    size_t node = i >> 7;        // /H
    int h = (int)(i & (H - 1));
    xcat[node * H4 + off + h] = x[i];
  }
}

// ---------------- attention score: A[n] = (a*b)[n,:] . wc + bc -------------
__global__ __launch_bounds__(128)
void attn_score_kernel(const float* __restrict__ ab, const float* __restrict__ wc,
                       const float* __restrict__ bc, float* __restrict__ Avec) {
  __shared__ float red[128];
  const int n = blockIdx.x, t = threadIdx.x;
  float s = 0.0f;
  for (int j = t; j < H4; j += 128) s += ab[(size_t)n * H4 + j] * wc[j];
  red[t] = s;
  __syncthreads();
  for (int st = 64; st > 0; st >>= 1) { if (t < st) red[t] += red[t + st]; __syncthreads(); }
  if (t == 0) Avec[n] = red[0] + bc[0];
}

// ---------------- softmax over all nodes (single block) ---------------------
__global__ __launch_bounds__(1024)
void node_softmax_kernel(const float* __restrict__ Avec, float* __restrict__ w, int n) {
  __shared__ float red[1024];
  const int t = threadIdx.x;
  float mx = -3.4e38f;
  for (int i = t; i < n; i += 1024) mx = fmaxf(mx, Avec[i]);
  red[t] = mx; __syncthreads();
  for (int s = 512; s > 0; s >>= 1) { if (t < s) red[t] = fmaxf(red[t], red[t + s]); __syncthreads(); }
  mx = red[0]; __syncthreads();
  float sm = 0.0f;
  for (int i = t; i < n; i += 1024) sm += __expf(Avec[i] - mx);
  red[t] = sm; __syncthreads();
  for (int s = 512; s > 0; s >>= 1) { if (t < s) red[t] += red[t + s]; __syncthreads(); }
  sm = red[0];
  for (int i = t; i < n; i += 1024) w[i] = __expf(Avec[i] - mx) / sm;
}

// ---------------- pooled[j] = sum_n w[n]*hp[n,j] ----------------------------
__global__ __launch_bounds__(256)
void pool_kernel(const float* __restrict__ w, const float* __restrict__ hp,
                 float* __restrict__ pooled, int n) {
  const int j = blockIdx.x * blockDim.x + threadIdx.x;   // 0..511
  float s = 0.0f;
  for (int i = 0; i < n; ++i) s += w[i] * hp[(size_t)i * H4 + j];
  pooled[j] = s;
}

// ---------------- head: out = relu(pooled@rho + rb) @ clf + cb --------------
__global__ __launch_bounds__(512)
void head_kernel(const float* __restrict__ pooled, const float* __restrict__ rho_w,
                 const float* __restrict__ rho_b, const float* __restrict__ clf_w,
                 const float* __restrict__ clf_b, float* __restrict__ out) {
  __shared__ float p[H4];
  __shared__ float r[H4];
  const int j = threadIdx.x;
  p[j] = pooled[j];
  __syncthreads();
  float s = rho_b[j];
  for (int k = 0; k < H4; ++k) s += p[k] * rho_w[(size_t)k * H4 + j];
  r[j] = fmaxf(s, 0.0f);
  __syncthreads();
  if (j < 3) {
    float o = clf_b[j];
    for (int k = 0; k < H4; ++k) o += r[k] * clf_w[(size_t)k * 3 + j];
    out[j] = o;
  }
}

}  // namespace

extern "C" void kernel_launch(void* const* d_in, const int* in_sizes, int n_in,
                              void* d_out, int out_size, void* d_ws, size_t ws_size,
                              hipStream_t stream) {
  (void)in_sizes; (void)n_in; (void)out_size; (void)ws_size;

  const float* features = (const float*)d_in[0];
  const int*   eidx     = (const int*)d_in[1];
  const float* ew       = (const float*)d_in[2];
  const float* fc_w     = (const float*)d_in[3];
  const float* fc_b     = (const float*)d_in[4];
  const float* conv_w1  = (const float*)d_in[5];
  const float* conv_b1  = (const float*)d_in[6];
  const float* conv_lng = (const float*)d_in[7];
  const float* conv_lnb = (const float*)d_in[8];
  const float* conv_w2  = (const float*)d_in[9];
  const float* conv_b2  = (const float*)d_in[10];
  const float* conv_t   = (const float*)d_in[11];
  const float* blk_lng  = (const float*)d_in[12];
  const float* blk_lnb  = (const float*)d_in[13];
  const float* phi_w    = (const float*)d_in[14];
  const float* phi_b    = (const float*)d_in[15];
  const float* attn_wa  = (const float*)d_in[16];
  const float* attn_ba  = (const float*)d_in[17];
  const float* attn_wb  = (const float*)d_in[18];
  const float* attn_bb  = (const float*)d_in[19];
  const float* attn_wc  = (const float*)d_in[20];
  const float* attn_bc  = (const float*)d_in[21];
  const float* rho_w    = (const float*)d_in[22];
  const float* rho_b    = (const float*)d_in[23];
  const float* clf_w    = (const float*)d_in[24];
  const float* clf_b    = (const float*)d_in[25];
  float* out = (float*)d_out;

  const int* src = eidx;
  const int* dst = eidx + E_EDGES;

  // ---- workspace carve (floats). Attention buffers alias conv scratch. ----
  const size_t NH  = (size_t)N_NODES * H;    // 2M
  const size_t NH2 = (size_t)N_NODES * H2;   // 4M
  const size_t NH4 = (size_t)N_NODES * H4;   // 8M
  float* ws = (float*)d_ws;
  float*    x    = ws;                       // [N,H]
  float*    xcat = x + NH;                   // [N,4H]
  float*    r0   = xcat + NH4;               // scratch arena (9*NH floats)
  unsigned* menc = (unsigned*)r0;            // [N,H]
  float*    se   = r0 + NH;                  // [N,H]
  float*    sme  = r0 + 2 * NH;              // [N,H]
  float*    hbuf = r0 + 3 * NH;              // [N,H]
  float*    t1   = r0 + 4 * NH;              // [N,2H]
  float*    t2   = r0 + 4 * NH + NH2;        // [N,2H]
  float*    hout = r0 + 4 * NH + 2 * NH2;    // [N,H]
  // attention phase (conv scratch dead by then):
  float* hp     = r0;                        // [N,4H]  (= menc..hbuf region)
  float* ab     = r0 + 4 * NH;               // [N,4H]  (= t1..t2 region)
  float* Avec   = hout;                      // [N]
  float* wvec   = hout + N_NODES;            // [N]
  float* pooled = hout + 2 * N_NODES;        // [4H]

  const int FB = 256;
  const int NHB = (int)(NH / FB);

  // ---- x = relu(features @ fc_w + fc_b) -----------------------------------
  wmma_gemm_kernel<1, false, DIN, H><<<dim3(H / 64, N_NODES / 64), 128, 0, stream>>>(
      features, fc_w, fc_b, x);
  concat_kernel<<<NHB, FB, 0, stream>>>(x, xcat, 0, NH);

  // ---- 3 GENConv layers ---------------------------------------------------
  for (int i = 0; i < 3; ++i) {
    // reset segment-softmax accumulators
    fill_u32_kernel<<<NHB, FB, 0, stream>>>(menc, 0x007FFFFFu /* ord(-inf) */, NH);
    fill_f32_kernel<<<NHB, FB, 0, stream>>>(se, 0.0f, NH);
    fill_f32_kernel<<<NHB, FB, 0, stream>>>(sme, 0.0f, NH);
    // edge passes
    edge_max_kernel<<<E_EDGES, H, 0, stream>>>(x, src, dst, ew, conv_t, i, menc);
    edge_sum_kernel<<<E_EDGES, H, 0, stream>>>(x, src, dst, ew, conv_t, i, menc, se, sme);
    // agg + root residual
    agg_kernel<<<NHB, FB, 0, stream>>>(se, sme, x, hbuf, NH);
    // MLP: Linear(H,2H) -> LN -> ReLU -> Linear(2H,H)
    wmma_gemm_kernel<0, false, H, H2><<<dim3(H2 / 64, N_NODES / 64), 128, 0, stream>>>(
        hbuf, conv_w1 + (size_t)i * H * H2, conv_b1 + (size_t)i * H2, t1);
    ln_kernel<H2, false><<<N_NODES, H2, 0, stream>>>(
        t1, conv_lng + (size_t)i * H2, conv_lnb + (size_t)i * H2, t2);
    float* gdst = (i == 0) ? x : hout;
    wmma_gemm_kernel<0, false, H2, H><<<dim3(H / 64, N_NODES / 64), 128, 0, stream>>>(
        t2, conv_w2 + (size_t)i * H2 * H, conv_b2 + (size_t)i * H, gdst);
    if (i > 0) {  // DeepGCN 'res': x = x + relu(LN(conv(x)))
      ln_kernel<H, true><<<N_NODES, H, 0, stream>>>(
          hout, blk_lng + (size_t)i * H, blk_lnb + (size_t)i * H, x);
    }
    concat_kernel<<<NHB, FB, 0, stream>>>(x, xcat, (i + 1) * H, NH);
  }

  // ---- attention pooling head --------------------------------------------
  wmma_gemm_kernel<1, false, H4, H4><<<dim3(H4 / 64, N_NODES / 64), 128, 0, stream>>>(
      xcat, phi_w, phi_b, hp);                                // hp = relu(...)
  wmma_gemm_kernel<2, false, H4, H4><<<dim3(H4 / 64, N_NODES / 64), 128, 0, stream>>>(
      hp, attn_wa, attn_ba, ab);                              // ab = tanh(...)
  wmma_gemm_kernel<3, true, H4, H4><<<dim3(H4 / 64, N_NODES / 64), 128, 0, stream>>>(
      hp, attn_wb, attn_bb, ab);                              // ab *= sigmoid(...)
  attn_score_kernel<<<N_NODES, 128, 0, stream>>>(ab, attn_wc, attn_bc, Avec);
  node_softmax_kernel<<<1, 1024, 0, stream>>>(Avec, wvec, N_NODES);
  pool_kernel<<<H4 / 256, 256, 0, stream>>>(wvec, hp, pooled, N_NODES);
  head_kernel<<<1, H4, 0, stream>>>(pooled, rho_w, rho_b, clf_w, clf_b, out);
}
